// VersorLinear_12146167513835
// MI455X (gfx1250) — matrix operand order repaired and checked by reading, not compiled, not run on hardware
//
#include <hip/hip_runtime.h>

// VersorLinear (Cl(4,1) geometric-product linear layer) as a single GEMM:
//   M = 4*512 = 2048, K = 256*32 = 8192, N = 256*32 = 8192
//   B[(i,l),(o,k)] = weight[o,i,l^k] * sign(l^k, l)   (generated on the fly in LDS)
// bf16x2 split (hi/lo) WMMA, fp32 accumulate, fused multivector L2-normalization.
// PRE path: x pre-split once into bf16 hi/lo planes in workspace; A tiles staged
// via CDNA5 async global->LDS DMA (ASYNCcnt), double-buffered LDS, 1 barrier/K-step.

typedef __attribute__((ext_vector_type(16))) __bf16 v16bf;
typedef __attribute__((ext_vector_type(8)))  float  v8f;
typedef int v4i __attribute__((vector_size(16)));      // matches builtin param type
typedef unsigned short u16;

#define M_DIM 2048
#define N_DIM 8192
#define K_DIM 8192
#define BLK_K 32
#define AP    40            // LDS pitch in bf16 elems (32 + 8 pad, keeps 16B align)
#define NT    (K_DIM / BLK_K)

#if __has_builtin(__builtin_amdgcn_global_load_async_to_lds_b128)
#define HAVE_ASYNC_LDS 1
#else
#define HAVE_ASYNC_LDS 0
#endif

static __device__ __forceinline__ u16 f2bf(float f) {
  unsigned u = __float_as_uint(f);
  u += 0x7FFFu + ((u >> 16) & 1u);            // round-to-nearest-even
  return (u16)(u >> 16);
}
static __device__ __forceinline__ float bf2f(u16 h) {
  return __uint_as_float(((unsigned)h) << 16);
}
// Sign of e_j * e_l in Cl(4,1): swap parity + metric of e4 (e4^2 = -1).
static __device__ __forceinline__ unsigned gp_signbit(unsigned j, unsigned l) {
  unsigned s = __popc((j >> 1) & l) + __popc((j >> 2) & l) +
               __popc((j >> 3) & l) + __popc((j >> 4) & l);
  s += (j & l) >> 4;
  return s & 1u;
}

union FragU { v16bf v; uint4 q[2]; };

// 16-byte global -> LDS copy; async DMA when the toolchain exposes it.
static __device__ __forceinline__ void copy16_g2l(const u16* g, u16* l) {
#if HAVE_ASYNC_LDS
  v4i* gp = (v4i*)g;                           // strip const, retype (generic AS)
  v4i* lp = (v4i*)l;
  __builtin_amdgcn_global_load_async_to_lds_b128(
      (__attribute__((address_space(1))) v4i*)gp,
      (__attribute__((address_space(3))) v4i*)lp, 0, 0);
#else
  *reinterpret_cast<uint4*>(l) = *reinterpret_cast<const uint4*>(g);
#endif
}
static __device__ __forceinline__ void wait_async_copies() {
#if HAVE_ASYNC_LDS
#if __has_builtin(__builtin_amdgcn_s_wait_asynccnt)
  __builtin_amdgcn_s_wait_asynccnt(0);
#else
  asm volatile("s_wait_asynccnt 0x0" ::: "memory");
#endif
#endif
}

// ---- one-shot x fp32 -> bf16 hi/lo plane split (bandwidth bound) ----
__global__ __launch_bounds__(256) void convert_x(const float* __restrict__ x,
                                                 u16* __restrict__ Xhi,
                                                 u16* __restrict__ Xlo) {
  size_t t  = (size_t)blockIdx.x * 256 + threadIdx.x;
  size_t i8 = t * 8;
  float4 f0 = *reinterpret_cast<const float4*>(x + i8);
  float4 f1 = *reinterpret_cast<const float4*>(x + i8 + 4);
  float f[8] = {f0.x, f0.y, f0.z, f0.w, f1.x, f1.y, f1.z, f1.w};
  union { uint4 q; u16 s[8]; } H, L;
  #pragma unroll
  for (int e = 0; e < 8; ++e) {
    H.s[e] = f2bf(f[e]);
    L.s[e] = f2bf(f[e] - bf2f(H.s[e]));
  }
  *reinterpret_cast<uint4*>(Xhi + i8) = H.q;
  *reinterpret_cast<uint4*>(Xlo + i8) = L.q;
}

template <bool PRE>
__global__ __launch_bounds__(256) void versor_gemm(const float* __restrict__ x,
                                                   const u16* __restrict__ Xhi,
                                                   const u16* __restrict__ Xlo,
                                                   const float* __restrict__ w,
                                                   float* __restrict__ out) {
  __shared__ alignas(16) u16 sAhi[2][128 * AP];
  __shared__ alignas(16) u16 sAlo[2][128 * AP];
  __shared__ alignas(16) u16 sBhi[2][128 * AP];
  __shared__ alignas(16) u16 sBlo[2][128 * AP];
  __shared__ float sW[2][128];    // weight row for one i: 4 o's x 32 blades

  const int tid  = threadIdx.x;
  const int lane = tid & 31;
  const int wave = tid >> 5;
  const int wm0  = (wave & 1) * 64;   // 2 x 4 wave grid, 64x32 per wave
  const int wn0  = (wave >> 1) * 32;
  const int m0   = blockIdx.y * 128;
  const int n0   = blockIdx.x * 128;
  const int o0   = blockIdx.x * 4;

  // B-tile generator mapping: thread -> (column n, half of K range)
  const int nrow    = tid >> 1;
  const int half    = tid & 1;
  const unsigned kb = (unsigned)(nrow & 31);
  const int olocal  = nrow >> 5;
  unsigned sw = 0;                    // bit l = sign of weight[.,.,l^kb] at blade kb
  #pragma unroll
  for (int l = 0; l < 32; ++l)
    sw |= gp_signbit((unsigned)l ^ kb, (unsigned)l) << l;

  v8f acc[4][2] = {};

  auto wIdx = [&](int kt) -> size_t {
    return ((size_t)(o0 + (tid >> 5)) * 256 + (size_t)kt) * 32 + (tid & 31);
  };

  auto stageA = [&](int st, int kt) {
    if constexpr (PRE) {
      // 4 x b128 async DMA per thread: 128 rows x 32 bf16, hi+lo planes.
      int row = tid >> 1;
      int cc  = (tid & 1) * 16;
      size_t g = (size_t)(m0 + row) * K_DIM + (size_t)kt * BLK_K + cc;
      u16* lh = &sAhi[st][row * AP + cc];
      u16* ll = &sAlo[st][row * AP + cc];
      copy16_g2l(Xhi + g,     lh);
      copy16_g2l(Xhi + g + 8, lh + 8);
      copy16_g2l(Xlo + g,     ll);
      copy16_g2l(Xlo + g + 8, ll + 8);
    } else {
      #pragma unroll
      for (int p = 0; p < 4; ++p) {
        int row = p * 32 + (tid >> 3);
        int col = (tid & 7) * 4;
        float4 a = *reinterpret_cast<const float4*>(
            x + (size_t)(m0 + row) * K_DIM + (size_t)kt * BLK_K + col);
        float f[4] = {a.x, a.y, a.z, a.w};
        u16* ph = &sAhi[st][row * AP + col];
        u16* pl = &sAlo[st][row * AP + col];
        #pragma unroll
        for (int e = 0; e < 4; e += 2) {
          u16 h0 = f2bf(f[e]), h1 = f2bf(f[e + 1]);
          *reinterpret_cast<unsigned*>(ph + e) = (unsigned)h0 | ((unsigned)h1 << 16);
          u16 q0 = f2bf(f[e] - bf2f(h0)), q1 = f2bf(f[e + 1] - bf2f(h1));
          *reinterpret_cast<unsigned*>(pl + e) = (unsigned)q0 | ((unsigned)q1 << 16);
        }
      }
    }
  };

  auto genB = [&](int st, int wb) {  // expand 128 weights -> 128x32 signed bf16 hi/lo
    const float* wrow = &sW[wb][olocal * 32];
    u16* bh = &sBhi[st][nrow * AP + half * 16];
    u16* bl = &sBlo[st][nrow * AP + half * 16];
    #pragma unroll
    for (int e = 0; e < 16; e += 2) {
      int l0 = half * 16 + e, l1 = l0 + 1;
      float f0 = wrow[(unsigned)l0 ^ kb];
      float f1 = wrow[(unsigned)l1 ^ kb];
      f0 = __uint_as_float(__float_as_uint(f0) ^ (((sw >> l0) & 1u) << 31));
      f1 = __uint_as_float(__float_as_uint(f1) ^ (((sw >> l1) & 1u) << 31));
      u16 h0 = f2bf(f0), h1 = f2bf(f1);
      *reinterpret_cast<unsigned*>(bh + e) = (unsigned)h0 | ((unsigned)h1 << 16);
      u16 q0 = f2bf(f0 - bf2f(h0)), q1 = f2bf(f1 - bf2f(h1));
      *reinterpret_cast<unsigned*>(bl + e) = (unsigned)q0 | ((unsigned)q1 << 16);
    }
  };

  // ISA A layout (16x32 bf16): lane<16 -> K {0-7,16-23}; lane>=16 -> K {8-15,24-31}.
  auto afrag = [&](const u16* plane, int row) -> v16bf {
    const u16* p = plane + row * AP + ((lane >> 4) * 8);
    FragU f;
    f.q[0] = *reinterpret_cast<const uint4*>(p);
    f.q[1] = *reinterpret_cast<const uint4*>(p + 16);
    return f.v;
  };
  // B layout (32x16 bf16): lane%16 = N column; lane<16 -> K {0-7,16-23}; lane>=16 -> K {8-15,24-31}.
  auto bfrag = [&](const u16* plane, int col) -> v16bf {
    const u16* p = plane + col * AP + ((lane >> 4) * 8);
    FragU f;
    f.q[0] = *reinterpret_cast<const uint4*>(p);
    f.q[1] = *reinterpret_cast<const uint4*>(p + 16);
    return f.v;
  };

  auto compute = [&](int st) {
    v16bf ah[4], al[4], bh[2], bl[2];
    #pragma unroll
    for (int mi = 0; mi < 4; ++mi) {
      int row = wm0 + mi * 16 + (lane & 15);
      ah[mi] = afrag(&sAhi[st][0], row);
      al[mi] = afrag(&sAlo[st][0], row);
    }
    #pragma unroll
    for (int ni = 0; ni < 2; ++ni) {
      int col = wn0 + ni * 16 + (lane & 15);
      bh[ni] = bfrag(&sBhi[st][0], col);
      bl[ni] = bfrag(&sBlo[st][0], col);
    }
    #pragma unroll
    for (int mi = 0; mi < 4; ++mi)
      #pragma unroll
      for (int ni = 0; ni < 2; ++ni) {
        acc[mi][ni] = __builtin_amdgcn_wmma_f32_16x16x32_bf16(
            false, ah[mi], false, bh[ni], (short)0, acc[mi][ni], false, false);
        acc[mi][ni] = __builtin_amdgcn_wmma_f32_16x16x32_bf16(
            false, ah[mi], false, bl[ni], (short)0, acc[mi][ni], false, false);
        acc[mi][ni] = __builtin_amdgcn_wmma_f32_16x16x32_bf16(
            false, al[mi], false, bh[ni], (short)0, acc[mi][ni], false, false);
      }
  };

  // ---- prologue: stage tile 0, prefetch weight rows 0 and 1 ----
  if (tid < 128) sW[0][tid] = w[wIdx(0)];
  stageA(0, 0);
  wait_async_copies();
  __syncthreads();                 // sW[0] + sA[0] visible
  genB(0, 0);
  if (tid < 128) sW[1][tid] = w[wIdx(1)];
  __syncthreads();                 // sB[0] + sW[1] visible

  // ---- main K loop: double-buffered, one barrier per K-step ----
  for (int kt = 0; kt < NT; ++kt) {
    const int cur = kt & 1, nxt = cur ^ 1;
    const bool more = (kt + 1) < NT;
    float wreg = 0.0f;
    if (more) {
      stageA(nxt, kt + 1);                 // async DMA overlaps the WMMAs below
      genB(nxt, (kt + 1) & 1);
      if ((kt + 2) < NT && tid < 128) wreg = w[wIdx(kt + 2)];
    }
    compute(cur);
    if (more) {
      if ((kt + 2) < NT && tid < 128) sW[kt & 1][tid] = wreg;
      wait_async_copies();
      __syncthreads();
    }
  }

  // ---- fused multivector normalization (32 coeffs = this wave's 2 N-subtiles) ----
  #pragma unroll
  for (int mi = 0; mi < 4; ++mi) {
    #pragma unroll
    for (int r = 0; r < 8; ++r) {
      float v0 = acc[mi][0][r], v1 = acc[mi][1][r];
      float ss = v0 * v0 + v1 * v1;
      ss += __shfl_xor(ss, 1);
      ss += __shfl_xor(ss, 2);
      ss += __shfl_xor(ss, 4);
      ss += __shfl_xor(ss, 8);       // reduce within 16-lane C-layout half
      float sc = rsqrtf(ss + 1e-6f);
      acc[mi][0][r] = v0 * sc;
      acc[mi][1][r] = v1 * sc;
    }
  }

  // ---- store (C layout: VGPR r -> row r, +8 for lanes 16-31; N = lane%16) ----
  const int colb = n0 + wn0 + (lane & 15);
  const int rowh = (lane >> 4) * 8;
  #pragma unroll
  for (int mi = 0; mi < 4; ++mi)
    #pragma unroll
    for (int ni = 0; ni < 2; ++ni)
      #pragma unroll
      for (int r = 0; r < 8; ++r) {
        int row = m0 + wm0 + mi * 16 + rowh + r;
        out[(size_t)row * N_DIM + (size_t)(colb + ni * 16)] = acc[mi][ni][r];
      }
}

extern "C" void kernel_launch(void* const* d_in, const int* in_sizes, int n_in,
                              void* d_out, int out_size, void* d_ws, size_t ws_size,
                              hipStream_t stream) {
  const float* x = (const float*)d_in[0];   // [4,512,256,32] fp32
  const float* w = (const float*)d_in[1];   // [256,256,32]  fp32
  float* out = (float*)d_out;               // [4,512,256,32] fp32

  const size_t planeElems = (size_t)M_DIM * K_DIM;
  u16* Xhi = (u16*)d_ws;
  u16* Xlo = Xhi + planeElems;

  dim3 grid(N_DIM / 128, M_DIM / 128);      // 64 x 16 blocks
  if (ws_size >= planeElems * 2 * sizeof(u16)) {
    convert_x<<<dim3((unsigned)(planeElems / 8 / 256)), dim3(256, 1, 1), 0, stream>>>(x, Xhi, Xlo);
    versor_gemm<true><<<grid, dim3(256, 1, 1), 0, stream>>>(x, Xhi, Xlo, w, out);
  } else {
    versor_gemm<false><<<grid, dim3(256, 1, 1), 0, stream>>>(x, Xhi, Xlo, w, out);
  }
  (void)in_sizes; (void)n_in; (void)out_size;
}
